// CoAttention_36567351558173
// MI455X (gfx1250) — matrix operand order repaired and checked by reading, not compiled, not run on hardware
//
#include <hip/hip_runtime.h>
#include <hip/hip_bf16.h>

// Problem constants (from reference): B=8192, QL=KVL=24, DQ=12, DKV=512
#define NB   8192
#define QL   24
#define KVL  24
#define DQ   12
#define DKV  512
#define E2   24          // 2*DQ
#define WAVES 4
#define TPB  (WAVES * 32)
#define BPW  2           // batches per wave: amortizes Wk LDS staging traffic
#define WKS  520         // padded bf16 row stride for Wk in LDS (bank de-conflict)

typedef __attribute__((ext_vector_type(16))) __bf16 v16bf;
typedef __attribute__((ext_vector_type(8)))  float  v8f;

__global__ __launch_bounds__(TPB) void coattn_kernel(
    const float* __restrict__ q,  const float* __restrict__ k,
    const float* __restrict__ Wq, const float* __restrict__ bq,
    const float* __restrict__ Wk, const float* __restrict__ bk,
    const float* __restrict__ Wl, const float* __restrict__ bl,
    float* __restrict__ out)
{
    // Block-shared weights
    __shared__ __attribute__((aligned(16))) __bf16 sWk[12 * WKS];   // Wk bf16, row = e (12), col = f (512)
    __shared__ float sWq[DQ * DQ];          // 144
    __shared__ float sWl[DQ * KVL];         // 288
    __shared__ float sbq[DQ], sbk[DQ], sbl[DQ];
    // Per-wave scratch (reused across BPW reps)
    __shared__ float sKp[WAVES][32 * 16];   // k_proj (padded 32x16)
    __shared__ float sQp[WAVES][QL * DQ];   // q_proj
    __shared__ float sQK[WAVES][QL * KVL];  // qk, then a_k in place
    __shared__ float sAq[WAVES][QL * KVL];  // a_q
    __shared__ float sQC[WAVES][QL * E2];   // [q | c_q]
    __shared__ float sCk[WAVES][KVL * E2];  // c_k

    const int tid  = threadIdx.x;
    const int wave = tid >> 5;
    const int lane = tid & 31;
    const int batch0 = blockIdx.x * (WAVES * BPW);   // block's first batch

    // Warm L2 for this wave's first k tile while weights are staged
    __builtin_prefetch(k + (size_t)(batch0 + wave) * KVL * DKV + lane * 16, 0, 1);

    // ---- Stage weights into LDS (cooperative, coalesced, once per block) ----
    for (int i = tid; i < 12 * DKV; i += TPB)
        sWk[(i / DKV) * WKS + (i % DKV)] = (__bf16)Wk[i];
    for (int i = tid; i < DQ * DQ;  i += TPB) sWq[i] = Wq[i];
    for (int i = tid; i < DQ * KVL; i += TPB) sWl[i] = Wl[i];
    if (tid < DQ) { sbq[tid] = bq[tid]; sbk[tid] = bk[tid]; sbl[tid] = bl[tid]; }
    __syncthreads();

    const int half = lane >> 4;              // 0: lanes 0-15, 1: lanes 16-31
    const int l16  = lane & 15;
    const int ec   = (l16 < DQ) ? l16 : (DQ - 1);   // clamped B column (N)

    for (int rep = 0; rep < BPW; ++rep) {
        const int batch = batch0 + rep * WAVES + wave;   // block's waves hit consecutive batches
        const float* kbase = k + (size_t)batch * KVL * DKV;
        const float* qbase = q + (size_t)batch * QL * DQ;

        // Prefetch next rep's k tile into L2 while this batch computes
        if (rep + 1 < BPW)
            __builtin_prefetch(kbase + (size_t)WAVES * KVL * DKV + lane * 16, 0, 1);

        // =================================================================
        // GEMM1 (dominant): k_proj[24,12] = k[24,512] @ Wk^T[512,12], WMMA
        // M padded to 32 (2 tiles), N padded to 16, K = 512 -> 16 steps
        // =================================================================
        v8f acc0 = {}; v8f acc1 = {};

        #pragma unroll
        for (int kk = 0; kk < DKV / 32; ++kk) {
            const int K0 = kk * 32;

            // B fragment (32x16 bf16): lane -> col N = l16, K = K0 + half*16 + i
            union { uint4 u[2]; v16bf v; } bf;
            const __bf16* wrow = &sWk[ec * WKS + K0 + half * 16];
            bf.u[0] = *(const uint4*)(wrow);
            bf.u[1] = *(const uint4*)(wrow + 8);

            // A fragment (16x32 bf16): lane row r, K = K0 + half*16 + {0..15}
            auto loadA = [&](int r) -> v16bf {
                const int rc = (r < KVL) ? r : (KVL - 1);   // clamp (padded rows unused)
                const float* arow = kbase + (size_t)rc * DKV + K0 + half * 16;
                float4 a0 = *(const float4*)(arow);
                float4 a1 = *(const float4*)(arow + 4);
                float4 a2 = *(const float4*)(arow + 8);
                float4 a3 = *(const float4*)(arow + 12);
                v16bf f;
                f[0]=(__bf16)a0.x;  f[1]=(__bf16)a0.y;  f[2]=(__bf16)a0.z;  f[3]=(__bf16)a0.w;
                f[4]=(__bf16)a1.x;  f[5]=(__bf16)a1.y;  f[6]=(__bf16)a1.z;  f[7]=(__bf16)a1.w;
                f[8]=(__bf16)a2.x;  f[9]=(__bf16)a2.y;  f[10]=(__bf16)a2.z; f[11]=(__bf16)a2.w;
                f[12]=(__bf16)a3.x; f[13]=(__bf16)a3.y; f[14]=(__bf16)a3.z; f[15]=(__bf16)a3.w;
                return f;
            };
            v16bf aT0 = loadA(l16);        // M tile 0: rows 0..15
            v16bf aT1 = loadA(16 + l16);   // M tile 1: rows 16..23 (+clamped pad)

            acc0 = __builtin_amdgcn_wmma_f32_16x16x32_bf16(false, aT0, false, bf.v,
                                                           (short)0, acc0, false, false);
            acc1 = __builtin_amdgcn_wmma_f32_16x16x32_bf16(false, aT1, false, bf.v,
                                                           (short)0, acc1, false, false);
        }

        // Spill k_proj (+bk) to per-wave LDS. C layout: lane -> N=l16, VGPR v -> M=v+half*8
        {
            const float bias = (l16 < DQ) ? sbk[l16] : 0.0f;
            #pragma unroll
            for (int v = 0; v < 8; ++v) {
                const int m = v + half * 8;
                sKp[wave][m * 16 + l16]        = acc0[v] + bias;
                sKp[wave][(16 + m) * 16 + l16] = acc1[v] + bias;
            }
        }
        __builtin_amdgcn_wave_barrier();

        // ---- q -> qc[:, 0:12] ----
        for (int i = lane; i < QL * DQ; i += 32) {
            const int qi = i / DQ, d = i % DQ;
            sQC[wave][qi * E2 + d] = qbase[i];
        }
        __builtin_amdgcn_wave_barrier();

        // ---- q_proj[q,e] = bq[e] + sum_d q[q,d]*Wq[e,d] ----
        for (int i = lane; i < QL * DQ; i += 32) {
            const int qi = i / DQ, e = i % DQ;
            float s = sbq[e];
            #pragma unroll
            for (int d = 0; d < DQ; ++d) s += sQC[wave][qi * E2 + d] * sWq[e * DQ + d];
            sQp[wave][qi * DQ + e] = s;
        }
        __builtin_amdgcn_wave_barrier();

        // ---- qk[q,k] = <q_proj[q,:], k_proj[k,:]> ----
        for (int i = lane; i < QL * KVL; i += 32) {
            const int qi = i / KVL, kj = i % KVL;
            float s = 0.0f;
            #pragma unroll
            for (int e = 0; e < DQ; ++e) s += sQp[wave][qi * DQ + e] * sKp[wave][kj * 16 + e];
            sQK[wave][i] = s;
        }
        __builtin_amdgcn_wave_barrier();

        // ---- a_q = softmax over q (columns) -> sAq ----
        if (lane < KVL) {
            const int kj = lane;
            float m = -3.402823466e38f;
            for (int qi = 0; qi < QL; ++qi) m = fmaxf(m, sQK[wave][qi * KVL + kj]);
            float ssum = 0.0f;
            for (int qi = 0; qi < QL; ++qi) {
                const float ex = __expf(sQK[wave][qi * KVL + kj] - m);
                sAq[wave][qi * KVL + kj] = ex; ssum += ex;
            }
            const float inv = 1.0f / ssum;
            for (int qi = 0; qi < QL; ++qi) sAq[wave][qi * KVL + kj] *= inv;
        }
        __builtin_amdgcn_wave_barrier();

        // ---- a_k = softmax over k (rows), in place in sQK ----
        if (lane < QL) {
            const int qi = lane;
            float m = -3.402823466e38f;
            for (int kj = 0; kj < KVL; ++kj) m = fmaxf(m, sQK[wave][qi * KVL + kj]);
            float ssum = 0.0f;
            for (int kj = 0; kj < KVL; ++kj) {
                const float ex = __expf(sQK[wave][qi * KVL + kj] - m);
                sQK[wave][qi * KVL + kj] = ex; ssum += ex;
            }
            const float inv = 1.0f / ssum;
            for (int kj = 0; kj < KVL; ++kj) sQK[wave][qi * KVL + kj] *= inv;
        }
        __builtin_amdgcn_wave_barrier();

        // ---- c_q[q,e] = sum_k a_q[q,k]*k_proj[k,e]  -> qc[:, 12:24] ----
        for (int i = lane; i < QL * DQ; i += 32) {
            const int qi = i / DQ, e = i % DQ;
            float s = 0.0f;
            #pragma unroll
            for (int kj = 0; kj < KVL; ++kj) s += sAq[wave][qi * KVL + kj] * sKp[wave][kj * 16 + e];
            sQC[wave][qi * E2 + DQ + e] = s;
        }
        __builtin_amdgcn_wave_barrier();

        // ---- c_k[k,e] = sum_q a_k[q,k]*qc[q,e] ----
        for (int i = lane; i < KVL * E2; i += 32) {
            const int kj = i / E2, e = i % E2;
            float s = 0.0f;
            #pragma unroll
            for (int qi = 0; qi < QL; ++qi) s += sQK[wave][qi * KVL + kj] * sQC[wave][qi * E2 + e];
            sCk[wave][kj * E2 + e] = s;
        }
        __builtin_amdgcn_wave_barrier();

        // ---- attn[e,d] = bl[d] + sum_k c_k[k,e]*Wl[d,k] ; out[b,e,d] ----
        for (int i = lane; i < E2 * DQ; i += 32) {
            const int e = i / DQ, d = i % DQ;
            float s = sbl[d];
            #pragma unroll
            for (int kj = 0; kj < KVL; ++kj) s += sCk[wave][kj * E2 + e] * sWl[d * KVL + kj];
            out[(size_t)batch * E2 * DQ + i] = s;
        }
        __builtin_amdgcn_wave_barrier();
    }
}

extern "C" void kernel_launch(void* const* d_in, const int* in_sizes, int n_in,
                              void* d_out, int out_size, void* d_ws, size_t ws_size,
                              hipStream_t stream) {
    const float* q  = (const float*)d_in[0];
    const float* k  = (const float*)d_in[1];
    // d_in[2] = v : unused by the reference; deliberately never read (halves HBM traffic)
    const float* Wq = (const float*)d_in[3];
    const float* bq = (const float*)d_in[4];
    const float* Wk = (const float*)d_in[5];
    const float* bk = (const float*)d_in[6];
    const float* Wl = (const float*)d_in[7];
    const float* bl = (const float*)d_in[8];
    float* out = (float*)d_out;

    dim3 grid(NB / (WAVES * BPW)), block(TPB);
    coattn_kernel<<<grid, block, 0, stream>>>(q, k, Wq, bq, Wk, bk, Wl, bl, out);
}
